// SplitOperation_41566693490840
// MI455X (gfx1250) — compile-verified, matrix operand order checked by reading, not run on hardware
//
#include <hip/hip_runtime.h>
#include <stdint.h>

typedef __attribute__((ext_vector_type(16))) _Float16 v16h;
typedef __attribute__((ext_vector_type(8)))  float    v8f;
typedef __attribute__((ext_vector_type(4)))  int      v4i;

// ---------------- gfx1250 async LDS path (probe-guarded) ----------------
#if defined(__has_builtin)
#if __has_builtin(__builtin_amdgcn_global_load_async_to_lds_b128) && \
    __has_builtin(__builtin_amdgcn_global_store_async_from_lds_b128)
#define USE_ASYNC_LDS 1
#endif
#endif
#ifndef USE_ASYNC_LDS
#define USE_ASYNC_LDS 0
#endif

#if USE_ASYNC_LDS
#if __has_builtin(__builtin_amdgcn_s_wait_asynccnt)
#define WAIT_ASYNC() __builtin_amdgcn_s_wait_asynccnt(0)
#else
#define WAIT_ASYNC() asm volatile("s_wait_asynccnt 0x0" ::: "memory")
#endif
// builtin signature (from probe diagnostics): (AS1 v4i*, AS3 v4i*, imm offset, imm cpol)
typedef __attribute__((address_space(1))) v4i* gv4i_p;
typedef __attribute__((address_space(3))) v4i* lv4i_p;
// flat shared-aperture address: low 32 bits are the LDS byte offset (ISA 10.2)
#define GPTR(p)   ((gv4i_p)(uintptr_t)(p))
#define LPTR(off) ((lv4i_p)(uint32_t)(off))
#endif

// ---------------- constants ----------------
#define A000 0.05103103630798287f   /* sqrt(1/384)  */
#define A12  0.08838834764831845f   /* sqrt(3/384)  */
#define IS3  0.57735026918962576f   /* 1/sqrt(3)    */
#define IS6  0.40824829046386302f   /* 1/sqrt(6)    */
#define RS8  0.35355339059327373f   /* 1/sqrt(8)    */

// workspace layout (bytes, all 32B-aligned)
#define WS_DESC0   0        /* 512  u32 */
#define WS_DESCK   2048     /* 3*768 u32 */
#define WS_B0      11264    /* 512*16 f16 (fragment-swizzled) */
#define WS_BK      27648    /* 768*16 f16 (fragment-swizzled) */

#define XSTR 68             /* padded row stride (floats) for staged X */
#define XIDX(r, c) ((r) * XSTR + (c))

// ---------------- descriptor / weight generation ----------------
__device__ __forceinline__ unsigned enc4(int a, int b, int c, int d, int m) {
  return (unsigned)(a | (b << 6) | (c << 12) | (d << 18) | (m << 24));
}

// feature p -> descriptor, out-irrep-0 tile (K = 512)
__device__ unsigned desc0_of(int p) {
  if (p < 256) { int u = p >> 4, v = p & 15; return enc4(u, v, 0, 0, 0); }
  if (p < 384) { int q = p - 256, pr = q >> 1, u = pr >> 3, v = pr & 7;
    int hb = 16 + 3 * u, ab = 16 + 3 * v;
    return (q & 1) ? enc4(hb + 2, ab + 2, 0, 0, 0)
                   : enc4(hb, ab, hb + 1, ab + 1, 1); }
  { int q = p - 384, pr = q >> 1, u = pr >> 3, v = pr & 7;
    int hb = 40 + 3 * u, ab = 40 + 3 * v;
    return (q & 1) ? enc4(hb + 2, ab + 2, 0, 0, 0)
                   : enc4(hb, ab, hb + 1, ab + 1, 1); }
}

// feature p -> descriptor, component-k tile (K = 768)
__device__ unsigned desck_of(int k, int p) {
  int k1 = (k + 1) % 3, k2 = (k + 2) % 3;
  if (p < 128) { int u = p >> 3, v = p & 7;  return enc4(u, 16 + 3 * v + k, 0, 0, 0); }
  if (p < 256) { int q = p - 128, u = q >> 4, v = q & 15; return enc4(16 + 3 * u + k, v, 0, 0, 0); }
  if (p < 320) { int q = p - 256, u = q >> 3, v = q & 7;
    return enc4(16 + 3 * u + k1, 40 + 3 * v + k2, 16 + 3 * u + k2, 40 + 3 * v + k1, 2); }
  if (p < 384) { int q = p - 320, u = q >> 3, v = q & 7;
    return enc4(40 + 3 * u + k1, 16 + 3 * v + k2, 40 + 3 * u + k2, 16 + 3 * v + k1, 2); }
  if (p < 512) { int q = p - 384, u = q >> 3, v = q & 7;  return enc4(u, 40 + 3 * v + k, 0, 0, 0); }
  if (p < 640) { int q = p - 512, u = q >> 4, v = q & 15; return enc4(40 + 3 * u + k, v, 0, 0, 0); }
  if (p < 704) { int q = p - 640, u = q >> 3, v = q & 7;
    return enc4(16 + 3 * u + k1, 16 + 3 * v + k2, 16 + 3 * u + k2, 16 + 3 * v + k1, 2); }
  { int q = p - 704, u = q >> 3, v = q & 7;
    return enc4(40 + 3 * u + k1, 40 + 3 * v + k2, 40 + 3 * u + k2, 40 + 3 * v + k1, 2); }
}

__device__ float b0_of(int p, int n, const float* t0, const float* t1, const float* t2) {
  if (p < 256) { int u = p >> 4, v = p & 15; return A000 * t0[(u * 16 + v) * 16 + n]; }
  if (p < 384) { int pr = (p - 256) >> 1, u = pr >> 3, v = pr & 7; return A000 * IS3 * t1[(u * 8 + v) * 16 + n]; }
  { int pr = (p - 384) >> 1, u = pr >> 3, v = pr & 7; return A000 * IS3 * t2[(u * 8 + v) * 16 + n]; }
}

__device__ float bk_of(int p, int n, const float* t3, const float* t4, const float* t5,
                       const float* t6, const float* t7, const float* t8,
                       const float* t9, const float* t10) {
  if (p < 128) { int u = p >> 3, v = p & 7;              return (n < 8) ? A12 * IS3 * t3[(u * 8 + v) * 8 + n] : 0.f; }
  if (p < 256) { int q = p - 128, u = q >> 4, v = q & 15; return (n < 8) ? A12 * IS3 * t4[(u * 16 + v) * 8 + n] : 0.f; }
  if (p < 320) { int q = p - 256, u = q >> 3, v = q & 7;  return (n < 8) ? A12 * IS6 * t5[(u * 8 + v) * 8 + n] : 0.f; }
  if (p < 384) { int q = p - 320, u = q >> 3, v = q & 7;  return (n < 8) ? A12 * IS6 * t6[(u * 8 + v) * 8 + n] : 0.f; }
  if (p < 512) { int q = p - 384, u = q >> 3, v = q & 7;  return (n >= 8) ? A12 * IS3 * t7[(u * 8 + v) * 8 + (n - 8)] : 0.f; }
  if (p < 640) { int q = p - 512, u = q >> 4, v = q & 15; return (n >= 8) ? A12 * IS3 * t8[(u * 16 + v) * 8 + (n - 8)] : 0.f; }
  if (p < 704) { int q = p - 640, u = q >> 3, v = q & 7;  return (n >= 8) ? A12 * IS6 * t9[(u * 8 + v) * 8 + (n - 8)] : 0.f; }
  { int q = p - 704, u = q >> 3, v = q & 7;               return (n >= 8) ? A12 * IS6 * t10[(u * 8 + v) * 8 + (n - 8)] : 0.f; }
}

// K index inside a 32-chunk for f16 A/B fragment slot e of lane L (wave32 layout)
__device__ __forceinline__ int kslot(int e, int half) {
  return (e & 7) + ((e >> 3) << 4) + (half << 3);
}

__global__ __launch_bounds__(256) void prep_kernel(
    const float* t0, const float* t1, const float* t2, const float* t3,
    const float* t4, const float* t5, const float* t6, const float* t7,
    const float* t8, const float* t9, const float* t10,
    uint32_t* desc0, uint32_t* desck, _Float16* b0, _Float16* bk) {
  const int t = threadIdx.x;
  // descriptors, pre-swizzled to A-fragment order [chunk][half][e]
  for (int s = t; s < 512; s += 256) {
    int c = s >> 5, r = s & 31, half = r >> 4, e = r & 15;
    desc0[s] = desc0_of(c * 32 + kslot(e, half));
  }
  for (int k = 0; k < 3; ++k)
    for (int s = t; s < 768; s += 256) {
      int c = s >> 5, r = s & 31, half = r >> 4, e = r & 15;
      desck[k * 768 + s] = desck_of(k, c * 32 + kslot(e, half));
    }
  // B matrices, pre-swizzled to per-lane fragment order [chunk][lane][e]
  for (int s = t; s < 512 * 16; s += 256) {
    int e = s & 15, L = (s >> 4) & 31, c = s >> 9;
    b0[s] = (_Float16)b0_of(c * 32 + kslot(e, L >> 4), L & 15, t0, t1, t2);
  }
  for (int s = t; s < 768 * 16; s += 256) {
    int e = s & 15, L = (s >> 4) & 31, c = s >> 9;
    bk[s] = (_Float16)bk_of(c * 32 + kslot(e, L >> 4), L & 15, t3, t4, t5, t6, t7, t8, t9, t10);
  }
}

// ---------------- main fused kernel ----------------
// 64 rows/block, 256 threads (8 waves).
// sx: staged X, row-major padded (stride 68) -> also reused as output staging.
// sh/sa: H and activated A, column-major [col][row] (conflict-free for FEAT).
__global__ __launch_bounds__(256) void e3nn_tp_kernel(
    const float* __restrict__ x,
    const float* __restrict__ lw0, const float* __restrict__ lw1, const float* __restrict__ lw2,
    const float* __restrict__ la0, const float* __restrict__ la1, const float* __restrict__ la2,
    const uint32_t* __restrict__ desc0, const uint32_t* __restrict__ desck,
    const _Float16* __restrict__ b0, const _Float16* __restrict__ bk,
    float* __restrict__ out) {
  __shared__ __attribute__((aligned(16))) float sx[64 * XSTR];
  __shared__ __attribute__((aligned(16))) float sh[64 * 64];
  __shared__ __attribute__((aligned(16))) float sa[64 * 64];

  const int t = threadIdx.x;
  const size_t base = (size_t)blockIdx.x * (64 * 64);

  // ---- phase 0: stage X -> LDS (row-major, padded) ----
#if USE_ASYNC_LDS
  const unsigned lds_sx = (unsigned)(uintptr_t)(void*)sx;
#pragma unroll
  for (int i = 0; i < 4; ++i) {
    int f = t + i * 256;
    int r = f >> 4, c0 = (f & 15) << 2;
    __builtin_amdgcn_global_load_async_to_lds_b128(
        GPTR(x + base + (size_t)f * 4), LPTR(lds_sx + (unsigned)(XIDX(r, c0) * 4)), 0, 0);
  }
  WAIT_ASYNC();
#else
#pragma unroll
  for (int i = 0; i < 4; ++i) {
    int f = t + i * 256;
    int r = f >> 4, c0 = (f & 15) << 2;
    float4 v = ((const float4*)(x + base))[f];
    *(float4*)(&sx[XIDX(r, c0)]) = v;
  }
#endif
  __syncthreads();

  // ---- phase 1: equivariant linears (fp32) + norm activation ----
  if (t < 128) {
    const int row = t & 63;
    const bool doA = (t >= 64);
    const float* W0 = doA ? la0 : lw0;
    const float* W1 = doA ? la1 : lw1;
    const float* W2 = doA ? la2 : lw2;
    float* dst = doA ? sa : sh;
    for (int v = 0; v < 16; ++v) {
      float acc = 0.f;
#pragma unroll
      for (int u = 0; u < 16; ++u) acc = fmaf(sx[XIDX(row, u)], W0[u * 16 + v], acc);
      dst[v * 64 + row] = acc * 0.25f;
    }
#pragma unroll
    for (int blk = 0; blk < 2; ++blk) {
      const int ib = 16 + blk * 24;
      const float* Wb = blk ? W2 : W1;
      for (int v = 0; v < 8; ++v)
#pragma unroll
        for (int i = 0; i < 3; ++i) {
          float acc = 0.f;
#pragma unroll
          for (int u = 0; u < 8; ++u)
            acc = fmaf(sx[XIDX(row, ib + u * 3 + i)], Wb[u * 8 + v], acc);
          dst[(ib + v * 3 + i) * 64 + row] = acc * RS8;
        }
    }
    if (doA) {  // NormActivation(sigmoid, normalize=True)
      for (int c = 0; c < 16; ++c) {
        float vv = sa[c * 64 + row];
        float nn = fabsf(vv);
        float s = 1.f / (1.f + expf(-nn));
        sa[c * 64 + row] = vv * s / (nn == 0.f ? 1.f : nn);
      }
      for (int g = 0; g < 16; ++g) {
        int cb = 16 + 3 * g;
        float a0 = sa[cb * 64 + row], a1 = sa[(cb + 1) * 64 + row], a2 = sa[(cb + 2) * 64 + row];
        float nn = sqrtf(a0 * a0 + a1 * a1 + a2 * a2);
        float s = 1.f / (1.f + expf(-nn));
        float fac = s / (nn == 0.f ? 1.f : nn);
        sa[cb * 64 + row] = a0 * fac;
        sa[(cb + 1) * 64 + row] = a1 * fac;
        sa[(cb + 2) * 64 + row] = a2 * fac;
      }
    }
  }
  __syncthreads();

  // ---- phase 2: tensor product as f16 WMMA matmuls ----
  const int lane = t & 31;
  const int half = lane >> 4;
  // force wave-uniform task selection into SGPRs so the WMMA chunk loop is a
  // scalar loop executed with full EXEC (ISA requirement for WMMA)
  const int wuni = __builtin_amdgcn_readfirstlane(t >> 5);
  float* so = sx;  // output staging aliases dead X buffer (row-major, stride 64)

#define FEAT(dw, ee)                                                         \
  {                                                                          \
    unsigned _d = (dw);                                                      \
    float _f = sh[(_d & 63u) * 64 + row] * sa[((_d >> 6) & 63u) * 64 + row]; \
    float _t2 = sh[((_d >> 12) & 63u) * 64 + row] *                          \
                sa[((_d >> 18) & 63u) * 64 + row];                           \
    int _m = (int)(_d >> 24);                                                \
    _f = fmaf((float)((_m == 1) - (_m == 2)), _t2, _f);                      \
    af[ee] = (_Float16)_f;                                                   \
  }

#pragma unroll
  for (int ti = 0; ti < 2; ++ti) {
    const int task = wuni * 2 + ti;
    const int rowblk = task >> 2, tile = task & 3;
    const int row = (rowblk << 4) | (lane & 15);
    const uint32_t* dT;
    const v16h* bT;
    int nch;
    if (tile == 0) { dT = desc0; bT = (const v16h*)b0; nch = 16; }
    else           { dT = desck + (tile - 1) * 768; bT = (const v16h*)bk; nch = 24; }

    v8f acc = {};
    for (int c = 0; c < nch; ++c) {
      const uint4* dq = (const uint4*)(dT + c * 32 + (half << 4));
      uint4 q0 = dq[0], q1 = dq[1], q2 = dq[2], q3 = dq[3];
      v16h af;
      FEAT(q0.x, 0)  FEAT(q0.y, 1)  FEAT(q0.z, 2)  FEAT(q0.w, 3)
      FEAT(q1.x, 4)  FEAT(q1.y, 5)  FEAT(q1.z, 6)  FEAT(q1.w, 7)
      FEAT(q2.x, 8)  FEAT(q2.y, 9)  FEAT(q2.z, 10) FEAT(q2.w, 11)
      FEAT(q3.x, 12) FEAT(q3.y, 13) FEAT(q3.z, 14) FEAT(q3.w, 15)
      v16h bf = bT[c * 32 + lane];
      acc = __builtin_amdgcn_wmma_f32_16x16x32_f16(
          false, af, false, bf, (short)0, acc, false, false);
    }
    const int n = lane & 15;
    const int col = (tile == 0) ? n : (16 + 3 * n + (tile - 1));
    const int orow = (rowblk << 4) + (half << 3);
#define STOREACC(rr) so[(orow + (rr)) * 64 + col] = acc[rr];
    STOREACC(0) STOREACC(1) STOREACC(2) STOREACC(3)
    STOREACC(4) STOREACC(5) STOREACC(6) STOREACC(7)
#undef STOREACC
  }
#undef FEAT
  __syncthreads();

  // ---- phase 3: coalesced output from LDS staging ----
#if USE_ASYNC_LDS
#pragma unroll
  for (int i = 0; i < 4; ++i) {
    int f = t + i * 256;
    __builtin_amdgcn_global_store_async_from_lds_b128(
        GPTR(out + base + (size_t)f * 4), LPTR(lds_sx + (unsigned)(f * 16)), 0, 0);
  }
  WAIT_ASYNC();
#else
#pragma unroll
  for (int i = 0; i < 4; ++i) {
    int f = t + i * 256;
    ((float4*)(out + base))[f] = ((const float4*)so)[f];
  }
#endif
}

// ---------------- host launcher ----------------
extern "C" void kernel_launch(void* const* d_in, const int* in_sizes, int n_in,
                              void* d_out, int out_size, void* d_ws, size_t ws_size,
                              hipStream_t stream) {
  const float* x = (const float*)d_in[0];
  const float* lw0 = (const float*)d_in[1];
  const float* lw1 = (const float*)d_in[2];
  const float* lw2 = (const float*)d_in[3];
  const float* la0 = (const float*)d_in[4];
  const float* la1 = (const float*)d_in[5];
  const float* la2 = (const float*)d_in[6];
  const float* tw[11];
  for (int i = 0; i < 11; ++i) tw[i] = (const float*)d_in[7 + i];

  char* ws = (char*)d_ws;
  uint32_t* desc0 = (uint32_t*)(ws + WS_DESC0);
  uint32_t* desck = (uint32_t*)(ws + WS_DESCK);
  _Float16* b0 = (_Float16*)(ws + WS_B0);
  _Float16* bk = (_Float16*)(ws + WS_BK);

  prep_kernel<<<1, 256, 0, stream>>>(tw[0], tw[1], tw[2], tw[3], tw[4], tw[5],
                                     tw[6], tw[7], tw[8], tw[9], tw[10],
                                     desc0, desck, b0, bk);

  const int nrow = in_sizes[0] / 64;
  const int nblocks = nrow / 64;
  e3nn_tp_kernel<<<nblocks, 256, 0, stream>>>(
      x, lw0, lw1, lw2, la0, la1, la2, desc0, desck, b0, bk, (float*)d_out);
}
